// StructureFeature_5076651344327
// MI455X (gfx1250) — matrix operand order repaired and checked by reading, not compiled, not run on hardware
//
#include <hip/hip_runtime.h>
#include <hip/hip_fp16.h>

typedef __attribute__((ext_vector_type(16))) _Float16 v16h;
typedef __attribute__((ext_vector_type(8)))  float    v8f;

#define Hc   256
#define Wc   512
#define Cc   32
#define HWc  131072    // 256*512 = 2^17
#define Pc   524288    // B*H*W   = 2^19
#define FEAT_ELEMS 8388608   // 4*16*H*W (out_feature region, floats)

// ---------------------------------------------------------------------------
// Kernel 1: L2-normalize over C, write packed f16 [p][32h]
// ---------------------------------------------------------------------------
__global__ void k_normalize(const float* __restrict__ x, __half* __restrict__ xn) {
    int p = blockIdx.x * blockDim.x + threadIdx.x;
    if (p >= Pc) return;
    int b = p >> 17;
    int rem = p & (HWc - 1);
    const float* xb = x + (size_t)b * Cc * HWc + rem;
    float v[32];
    float ss = 0.f;
#pragma unroll
    for (int c = 0; c < 32; ++c) {
        v[c] = xb[(size_t)c * HWc];
        ss = fmaf(v[c], v[c], ss);
    }
    float inv = 1.0f / fmaxf(sqrtf(ss), 1e-12f);
    union { uint4 u[4]; __half2 h[16]; } o;
#pragma unroll
    for (int j = 0; j < 16; ++j)
        o.h[j] = __floats2half2_rn(v[2 * j] * inv, v[2 * j + 1] * inv);
    uint4* dst = (uint4*)(xn + (size_t)p * 32);
    dst[0] = o.u[0]; dst[1] = o.u[1]; dst[2] = o.u[2]; dst[3] = o.u[3];
}

// ---------------------------------------------------------------------------
// Kernel 2: affinity for 4 dilations x 8 offsets; f32 to d_out, f16 to ws.
// Neighbor displacement is a compile-time constant per unrolled (di,k), so
// all 32 neighbor loads use immediate IOFFSETs off one per-pixel base.
// ---------------------------------------------------------------------------
__global__ void k_affinity(const __half* __restrict__ xn,
                           float* __restrict__ aff_f,
                           __half* __restrict__ aff_h) {
    const int tid = threadIdx.x;
    const int w = blockIdx.x * 64 + (tid & 63);
    const int h = blockIdx.y * 4 + (tid >> 6);
    const int b = blockIdx.z;
    const int p = (b << 17) + h * Wc + w;

    const uint4* cp = (const uint4*)(xn + (size_t)p * 32);   // 4 uint4 per pixel
    union { uint4 u[4]; __half2 t[16]; } cen;
    cen.u[0] = cp[0]; cen.u[1] = cp[1]; cen.u[2] = cp[2]; cen.u[3] = cp[3];

    const int dh[8] = {-1, -1, -1, 0, 0, 1, 1, 1};
    const int dw[8] = {-1, 0, 1, -1, 1, -1, 0, 1};
    __half oh[32];

#pragma unroll
    for (int di = 0; di < 4; ++di) {
        const int d = 1 << di;
#pragma unroll
        for (int k = 0; k < 8; ++k) {
            const int hh = h + dh[k] * d;
            const int ww = w + dw[k] * d;
            float a = 0.f;
            if (hh >= 0 && hh < Hc && ww >= 0 && ww < Wc) {
                // constant displacement (pixels) -> immediate byte offset
                const int disp = (dh[k] * Wc + dw[k]) * d * 4;   // uint4 units
                union { uint4 u[4]; __half2 t[16]; } nb;
                nb.u[0] = cp[disp + 0];
                nb.u[1] = cp[disp + 1];
                nb.u[2] = cp[disp + 2];
                nb.u[3] = cp[disp + 3];
                float a0 = 0.f, a1 = 0.f;   // f32 accum from f16 srcs -> v_fma_mix
#pragma unroll
                for (int j = 0; j < 16; ++j) {
                    a0 = fmaf(__low2float(cen.t[j]),  __low2float(nb.t[j]),  a0);
                    a1 = fmaf(__high2float(cen.t[j]), __high2float(nb.t[j]), a1);
                }
                a = fmaxf(a0 + a1, 0.f);
            }
            const int ch = di * 8 + k;
            aff_f[((size_t)(b * 32 + ch) << 17) + h * Wc + w] = a;
            oh[ch] = __float2half_rn(a);
        }
    }
    union { uint4 u[4]; __half hh[32]; } pk;
#pragma unroll
    for (int c = 0; c < 32; ++c) pk.hh[c] = oh[c];
    uint4* op = (uint4*)(aff_h + (size_t)p * 32);
    op[0] = pk.u[0]; op[1] = pk.u[1]; op[2] = pk.u[2]; op[3] = pk.u[3];
}

// ---------------------------------------------------------------------------
// WMMA helpers: A = 16x32 block-diagonal W' (f16), B = aff columns (f16)
// A layout: lanes<16 hold K 0..15; lanes>=16 hold K 16..31 (contiguous, matches B)
// ---------------------------------------------------------------------------
__device__ __forceinline__ v16h build_A(const float* __restrict__ Wf, int lane) {
    v16h a;
    const int m  = lane & 15;       // output channel row (i*4+o)
    const int hi = lane >> 4;
#pragma unroll
    for (int t = 0; t < 16; ++t) {
        int k  = t + (hi ? 16 : 0);  // K index 0..31
        int i2 = k >> 3;                                 // dilation of K
        float val = ((m >> 2) == i2) ? Wf[i2 * 32 + (m & 3) * 8 + (k & 7)] : 0.f;
        a[t] = (_Float16)val;
    }
    return a;
}

// Per-lane base for this wave's B tiles: group g is at +g*64 uint4 (1024 B).
__device__ __forceinline__ const uint4* B_base(const __half* __restrict__ aff_h,
                                               int gbase, int lane) {
    const int px = gbase * 16 + (lane & 15);
    return (const uint4*)(aff_h + (size_t)px * 32 + (lane >> 4) * 16);
}

__device__ __forceinline__ v16h load_B(const uint4* __restrict__ base, int g) {
    union { uint4 u[2]; v16h h; } bb;
    bb.u[0] = base[g * 64 + 0];     // immediate offsets under full unroll
    bb.u[1] = base[g * 64 + 1];
    return bb.h;
}

// ---------------------------------------------------------------------------
// Kernel 3: WMMA conv pass #1 — accumulate per-channel sum / sumsq of y
// ---------------------------------------------------------------------------
__global__ void k_conv_stats(const __half* __restrict__ aff_h,
                             const float* __restrict__ Wf,
                             float* __restrict__ gstats) {
    __shared__ float s_acc[32];
    const int tid  = threadIdx.x;
    const int lane = tid & 31;
    const int wv   = tid >> 5;
    if (tid < 32) s_acc[tid] = 0.f;
    __syncthreads();

    const v16h A = build_A(Wf, lane);
    float sum[8], sq[8];
#pragma unroll
    for (int r = 0; r < 8; ++r) { sum[r] = 0.f; sq[r] = 0.f; }

    const int gbase = (blockIdx.x * 8 + wv) * 16;   // 16 pixel-groups per wave
    const uint4* base = B_base(aff_h, gbase, lane);
#pragma unroll
    for (int g = 0; g < 16; ++g) {
        v16h B = load_B(base, g);
        v8f  C = {};
        C = __builtin_amdgcn_wmma_f32_16x16x32_f16(false, A, false, B,
                                                   (short)0, C, false, false);
#pragma unroll
        for (int r = 0; r < 8; ++r) {
            sum[r] += C[r];
            sq[r]   = fmaf(C[r], C[r], sq[r]);
        }
    }
    // butterfly reduce within each 16-lane half (channels 0-7 / 8-15)
#pragma unroll
    for (int m = 8; m >= 1; m >>= 1) {
#pragma unroll
        for (int r = 0; r < 8; ++r) {
            sum[r] += __shfl_xor(sum[r], m, 32);
            sq[r]  += __shfl_xor(sq[r],  m, 32);
        }
    }
    if ((lane & 15) == 0) {
        const int ch0 = (lane >> 4) * 8;
#pragma unroll
        for (int r = 0; r < 8; ++r) {
            atomicAdd(&s_acc[ch0 + r],      sum[r]);
            atomicAdd(&s_acc[16 + ch0 + r], sq[r]);
        }
    }
    __syncthreads();
    if (tid < 32) atomicAdd(&gstats[tid], s_acc[tid]);
}

// ---------------------------------------------------------------------------
// Kernel 4: fold BN stats into per-channel scale/shift
// ---------------------------------------------------------------------------
__global__ void k_bn(float* __restrict__ stats,
                     const float* __restrict__ gamma,
                     const float* __restrict__ beta) {
    const int ch = threadIdx.x;          // 16 threads
    const float N = (float)Pc;
    float mu  = stats[ch] / N;
    float var = stats[16 + ch] / N - mu * mu;
    float sc  = gamma[ch] * rsqrtf(var + 1e-5f);
    stats[32 + ch] = sc;
    stats[48 + ch] = beta[ch] - mu * sc;
}

// ---------------------------------------------------------------------------
// Kernel 5: WMMA conv pass #2 — apply BN affine + ReLU, write out_feature.
// A wave's 256-pixel range is 256-aligned -> never crosses a batch boundary,
// so the store base is loop-invariant and offsets are immediates.
// ---------------------------------------------------------------------------
__global__ void k_conv_apply(const __half* __restrict__ aff_h,
                             const float* __restrict__ Wf,
                             const float* __restrict__ stats,
                             float* __restrict__ out) {
    const int tid  = threadIdx.x;
    const int lane = tid & 31;
    const int wv   = tid >> 5;
    const v16h A = build_A(Wf, lane);
    const int ch0 = (lane >> 4) * 8;
    float sc[8], sh[8];
#pragma unroll
    for (int r = 0; r < 8; ++r) {
        sc[r] = stats[32 + ch0 + r];
        sh[r] = stats[48 + ch0 + r];
    }
    const int gbase = (blockIdx.x * 8 + wv) * 16;
    const uint4* base = B_base(aff_h, gbase, lane);

    const int p0   = gbase * 16 + (lane & 15);     // first pixel of this lane
    const int b    = p0 >> 17;
    const int rem0 = p0 & (HWc - 1);
    float* obase = out + ((size_t)(b * 16 + ch0) << 17) + rem0;

#pragma unroll
    for (int g = 0; g < 16; ++g) {
        v16h B = load_B(base, g);
        v8f  C = {};
        C = __builtin_amdgcn_wmma_f32_16x16x32_f16(false, A, false, B,
                                                   (short)0, C, false, false);
#pragma unroll
        for (int r = 0; r < 8; ++r)
            obase[((size_t)r << 17) + g * 16] = fmaxf(fmaf(C[r], sc[r], sh[r]), 0.f);
    }
}

// ---------------------------------------------------------------------------
extern "C" void kernel_launch(void* const* d_in, const int* in_sizes, int n_in,
                              void* d_out, int out_size, void* d_ws, size_t ws_size,
                              hipStream_t stream) {
    (void)in_sizes; (void)n_in; (void)out_size; (void)ws_size;
    const float* x     = (const float*)d_in[0];   // [4,32,256,512]
    const float* Wf    = (const float*)d_in[1];   // [4,4,8]
    const float* gamma = (const float*)d_in[2];   // [4,4]
    const float* beta  = (const float*)d_in[3];   // [4,4]

    float* out   = (float*)d_out;
    float* aff_f = out + FEAT_ELEMS;              // affinity region [4,32,H,W]
    __half* xn   = (__half*)out;                  // stage xn-f16 in out_feature region (dead until k5)
    __half* aff_h = (__half*)d_ws;                // 33,554,432 B: aff f16 [p][32]
    float* stats = (float*)((char*)d_ws + 33554432); // 64 floats: sum,sq,scale,shift

    hipMemsetAsync(stats, 0, 64 * sizeof(float), stream);

    k_normalize<<<Pc / 256, 256, 0, stream>>>(x, xn);
    k_affinity<<<dim3(Wc / 64, Hc / 4, 4), 256, 0, stream>>>(xn, aff_f, aff_h);
    k_conv_stats<<<256, 256, 0, stream>>>(aff_h, Wf, stats);   // 256*8 waves *16 groups *16 px = Pc
    k_bn<<<1, 16, 0, stream>>>(stats, gamma, beta);
    k_conv_apply<<<256, 256, 0, stream>>>(aff_h, Wf, stats, out);
}
